// V2PNet_67929202753918
// MI455X (gfx1250) — matrix operand order repaired and checked by reading, not compiled, not run on hardware
//
#include <hip/hip_runtime.h>

// ---------------------------------------------------------------------------
// Sparse-conv residual block (MinkowskiNet style) for gfx1250.
//   N=100000 voxels, K=27 offsets, C=128 channels.
//   conv -> BN -> ReLU -> conv -> BN -> (+x) -> ReLU
// bf16 WMMA (v_wmma_f32_16x16x32_bf16) with fp32 accumulation.
// Pipeline per block:
//   - double-buffered 34KB LDS activation tiles
//   - GLOBAL_LOAD_ASYNC_TO_LDS_B128 gather for k+1 overlaps WMMAs for k
//   - one s_barrier per k iteration, A-tiles prefetched 2 subtiles deep
// ---------------------------------------------------------------------------

#define NV    100000
#define KOFF  27
#define CH    128
#define MT    128     // voxel rows per block
#define PITCH 136     // LDS row pitch in bf16 elems (272 B = 68 dwords -> 4-bank lane stride)

typedef __attribute__((ext_vector_type(16))) __bf16 v16bf;
typedef __attribute__((ext_vector_type(8)))  __bf16 v8bf;
typedef __attribute__((ext_vector_type(8)))  float  v8f;

#if __has_builtin(__builtin_amdgcn_global_load_async_to_lds_b128)
#define ASYNC_GATHER 1
// exact parameter types per the compiler diagnostic: int4 in AS1 / AS3
typedef int v4i __attribute__((vector_size(16)));
typedef __attribute__((address_space(1))) v4i* g_v4ip;   // global int4*
typedef __attribute__((address_space(3))) v4i* l_v4ip;   // LDS int4*
#else
#define ASYNC_GATHER 0
#endif

#if ASYNC_GATHER
#if __has_builtin(__builtin_amdgcn_s_wait_asynccnt)
#define WAIT_ASYNC0() __builtin_amdgcn_s_wait_asynccnt(0)
#else
#define WAIT_ASYNC0() asm volatile("s_wait_asynccnt 0x0" ::: "memory")
#endif
#else
#define WAIT_ASYNC0() ((void)0)
#endif

__device__ __forceinline__ unsigned short f2bf(float f) {
  union { float f; unsigned u; } c; c.f = f;
  unsigned r = c.u + 0x7FFFu + ((c.u >> 16) & 1u);   // round to nearest even
  return (unsigned short)(r >> 16);
}

// --- fp32 -> bf16 conversion of activations --------------------------------
__global__ void cvt_x_bf16(const float* __restrict__ src,
                           unsigned short* __restrict__ dst, int n) {
  int i = blockIdx.x * blockDim.x + threadIdx.x;
  if (i < n) dst[i] = f2bf(src[i]);
}

// --- pack W[k] (C_in x C_out, row major) into WMMA B lane layout -----------
// Wp[((k*8 + n16)*4 + t)*32 + lane][j]  (16 bf16 each, contiguous per lane)
//   lane<16 holds K = t*32 + j,  lane>=16 holds K = t*32 + 16 + j,
//   column n = n16*16 + (lane&15).
__global__ void pack_w(const float* __restrict__ W,
                       unsigned short* __restrict__ Wp) {
  int id = blockIdx.x * blockDim.x + threadIdx.x;
  if (id >= KOFF * 1024) return;
  int lane = id & 31;
  int t    = (id >> 5) & 3;
  int n16  = (id >> 7) & 7;
  int k    = id >> 10;
  int n  = n16 * 16 + (lane & 15);
  int kb = t * 32 + ((lane & 16) ? 16 : 0);
  unsigned short* dst = Wp + (size_t)id * 16;
#pragma unroll
  for (int j = 0; j < 16; ++j)
    dst[j] = f2bf(W[((size_t)k * CH + (kb + j)) * CH + n]);
}

// --- sparse conv: out[i] = sum_k in[nbr[i,k]] @ W[k] -----------------------
__global__ __launch_bounds__(256)
void spconv(const unsigned short* __restrict__ xb,   // [NV][CH] bf16
            const int*            __restrict__ nbr,  // [NV][KOFF]
            const unsigned short* __restrict__ Wp,   // packed bf16 weights
            float*                __restrict__ hout, // [NV][CH] fp32
            float*                __restrict__ psum, // [nblk][CH] partial sums
            float*                __restrict__ psq)  // [nblk][CH] partial sumsq
{
  __shared__ unsigned short smem[2][MT * PITCH];     // 2 x 34 KB double buffer
  const int tid  = threadIdx.x;
  const int lane = tid & 31;
  const int wav  = tid >> 5;                         // 0..7 -> 16-col group
  const int row0 = blockIdx.x * MT;

  v8f acc[8];
#pragma unroll
  for (int s = 0; s < 8; ++s)
#pragma unroll
    for (int r = 0; r < 8; ++r) acc[s][r] = 0.0f;

  // gather assignment: 2 threads per row, 128 B each
  const int grow   = tid >> 1;
  const int ghalf  = (tid & 1) * 64;                 // bf16 element offset
  const int gvox   = row0 + grow;
  const bool gvld  = gvox < NV;

  const int lrow = lane & 15;
  const int chi  = (lane & 16) ? 16 : 0;             // A sub-column per half-wave

  union AV { v16bf v; v8bf h[2]; };

  // issue the (per-lane DMA) gather of kernel-offset k into LDS buffer `buf`
  auto issue_gather = [&](int k, int buf) {
    if (!gvld) return;                               // pad rows pre-zeroed
    unsigned short* sdst = &smem[buf][grow * PITCH + ghalf];
    const unsigned short* srcp =
        xb + (size_t)nbr[gvox * KOFF + k] * CH + ghalf;
#if ASYNC_GATHER
#pragma unroll
    for (int j = 0; j < 8; ++j)
      __builtin_amdgcn_global_load_async_to_lds_b128(
          (g_v4ip)(unsigned long long)(srcp + j * 8),
          (l_v4ip)(sdst + j * 8), 0, 0);
#else
#pragma unroll
    for (int j = 0; j < 8; ++j)
      ((uint4*)sdst)[j] = ((const uint4*)srcp)[j];
#endif
  };

  auto load_B = [&](int k, v16bf* B) {
    const unsigned short* wp =
        Wp + ((size_t)((k * 8 + wav) * 4) * 32 + lane) * 16;
#pragma unroll
    for (int t = 0; t < 4; ++t)
      B[t] = *(const v16bf*)(wp + (size_t)t * 32 * 16);
  };

  // pad rows: fill once in both buffers (tail block only)
  if (!gvld) {
    uint4 z; z.x = z.y = z.z = z.w = 0u;
#pragma unroll
    for (int b = 0; b < 2; ++b) {
      unsigned short* sdst = &smem[b][grow * PITCH + ghalf];
#pragma unroll
      for (int j = 0; j < 8; ++j) ((uint4*)sdst)[j] = z;
    }
  }

  // ---- prologue: gather(0) + B(0) ----
  issue_gather(0, 0);
  v16bf B[4];
  load_B(0, B);
  WAIT_ASYNC0();
  __syncthreads();

  for (int k = 0; k < KOFF; ++k) {
    const int cur = k & 1;
    if (k + 1 < KOFF) {
      issue_gather(k + 1, cur ^ 1);                  // DMA overlaps WMMAs below
      __builtin_prefetch(                            // global_prefetch_b8: W(k+2)
          Wp + ((size_t)(((k + 2) * 8 + wav) * 4) * 32 + lane) * 16, 0, 0);
    }

    // ---- 4 K-steps x 8 row-subtiles; A prefetched 2 subtiles deep ----
    const unsigned short* sbase = &smem[cur][0];
#pragma unroll
    for (int t = 0; t < 4; ++t) {
      AV A[8];
#pragma unroll
      for (int p = 0; p < 2; ++p) {                  // preload s=0,1
        const unsigned short* ar = sbase + (p * 16 + lrow) * PITCH + t * 32 + chi;
        A[p].h[0] = *(const v8bf*)(ar);
        A[p].h[1] = *(const v8bf*)(ar + 8);
      }
#pragma unroll
      for (int s = 0; s < 8; ++s) {
        if (s + 2 < 8) {                             // prefetch subtile s+2
          const unsigned short* ar =
              sbase + ((s + 2) * 16 + lrow) * PITCH + t * 32 + chi;
          A[s + 2].h[0] = *(const v8bf*)(ar);
          A[s + 2].h[1] = *(const v8bf*)(ar + 8);
        }
        acc[s] = __builtin_amdgcn_wmma_f32_16x16x32_bf16(
            false, A[s].v, false, B[t], (short)0, acc[s], false, false);
      }
    }

    if (k + 1 < KOFF) load_B(k + 1, B);              // latency hidden by barrier
    WAIT_ASYNC0();                                   // this wave's gather(k+1) done
    __syncthreads();                                 // buf(cur) free, buf(nxt) full
  }

  // ---- write fp32 result + deterministic per-block BN partials ----
  float cs = 0.0f, cq = 0.0f;
  const int colg = wav * 16 + lrow;
  const int rb   = row0 + ((lane & 16) ? 8 : 0);
#pragma unroll
  for (int s = 0; s < 8; ++s)
#pragma unroll
    for (int r = 0; r < 8; ++r) {
      float v = acc[s][r];
      int row = rb + s * 16 + r;
      if (row < NV) hout[(size_t)row * CH + colg] = v;
      cs += v; cq += v * v;                           // pad rows contribute 0
    }
  cs += __shfl_xor(cs, 16);                           // fold M and M+8 halves
  cq += __shfl_xor(cq, 16);
  if (lane < 16) {
    psum[(size_t)blockIdx.x * CH + colg] = cs;
    psq [(size_t)blockIdx.x * CH + colg] = cq;
  }
}

// --- fold per-block partials into per-channel sums (deterministic) ---------
__global__ void reduce_stats(const float* __restrict__ psum,
                             const float* __restrict__ psq,
                             float* __restrict__ ssum, float* __restrict__ ssq,
                             int nblk) {
  int c = threadIdx.x;                                // 128 threads
  float s = 0.0f, q = 0.0f;
  for (int b = 0; b < nblk; ++b) {
    s += psum[(size_t)b * CH + c];
    q += psq [(size_t)b * CH + c];
  }
  ssum[c] = s; ssq[c] = q;
}

// --- BN + ReLU -> bf16 (layer 1 epilogue) ----------------------------------
__global__ void bn_relu_bf16(const float* __restrict__ h,
                             const float* __restrict__ ssum,
                             const float* __restrict__ ssq,
                             const float* __restrict__ gamma,
                             const float* __restrict__ beta,
                             unsigned short* __restrict__ out) {
  int i = blockIdx.x * blockDim.x + threadIdx.x;      // exact grid: NV*CH
  int c = i & (CH - 1);
  float mean = ssum[c] * (1.0f / NV);
  float var  = ssq [c] * (1.0f / NV) - mean * mean;
  float sc = gamma[c] * rsqrtf(var + 1e-5f);
  float sh = beta[c] - mean * sc;
  float v  = fmaf(h[i], sc, sh);
  out[i] = f2bf(fmaxf(v, 0.0f));
}

// --- BN + residual + ReLU -> fp32 (final epilogue) -------------------------
__global__ void bn_res_relu(const float* __restrict__ h,
                            const float* __restrict__ ssum,
                            const float* __restrict__ ssq,
                            const float* __restrict__ gamma,
                            const float* __restrict__ beta,
                            const float* __restrict__ x,
                            float* __restrict__ out) {
  int i = blockIdx.x * blockDim.x + threadIdx.x;      // exact grid: NV*CH
  int c = i & (CH - 1);
  float mean = ssum[c] * (1.0f / NV);
  float var  = ssq [c] * (1.0f / NV) - mean * mean;
  float sc = gamma[c] * rsqrtf(var + 1e-5f);
  float sh = beta[c] - mean * sc;
  float v  = fmaf(h[i], sc, sh) + x[i];
  out[i] = fmaxf(v, 0.0f);
}

// ---------------------------------------------------------------------------
extern "C" void kernel_launch(void* const* d_in, const int* in_sizes, int n_in,
                              void* d_out, int out_size, void* d_ws, size_t ws_size,
                              hipStream_t stream) {
  const float* x  = (const float*)d_in[0];
  const float* W1 = (const float*)d_in[1];
  const float* W2 = (const float*)d_in[2];
  const float* g1 = (const float*)d_in[3];
  const float* b1 = (const float*)d_in[4];
  const float* g2 = (const float*)d_in[5];
  const float* b2 = (const float*)d_in[6];
  const int*   n1 = (const int*)d_in[7];
  const int*   n2 = (const int*)d_in[8];
  float* out = (float*)d_out;

  const int NBLK = (NV + MT - 1) / MT;                // 782

  char* ws = (char*)d_ws;
  size_t off = 0;
  auto alloc = [&](size_t bytes) {
    void* p = ws + off;
    off = (off + bytes + 255) & ~(size_t)255;
    return p;
  };
  unsigned short* xb  = (unsigned short*)alloc((size_t)NV * CH * 2);
  unsigned short* h1b = (unsigned short*)alloc((size_t)NV * CH * 2);
  unsigned short* W1p = (unsigned short*)alloc((size_t)KOFF * CH * CH * 2);
  unsigned short* W2p = (unsigned short*)alloc((size_t)KOFF * CH * CH * 2);
  float* hf   = (float*)alloc((size_t)NV * CH * 4);
  float* psum = (float*)alloc((size_t)NBLK * CH * 4);
  float* psq  = (float*)alloc((size_t)NBLK * CH * 4);
  float* ssum = (float*)alloc(CH * 4);
  float* ssq  = (float*)alloc(CH * 4);

  const int elemGrid = (NV * CH) / 256;               // 50000, exact
  const int packGrid = (KOFF * 1024 + 255) / 256;     // 108

  cvt_x_bf16 <<<elemGrid, 256, 0, stream>>>(x, xb, NV * CH);
  pack_w     <<<packGrid, 256, 0, stream>>>(W1, W1p);
  pack_w     <<<packGrid, 256, 0, stream>>>(W2, W2p);

  // layer 1: conv -> BN stats -> BN+ReLU (bf16)
  spconv       <<<NBLK, 256, 0, stream>>>(xb, n1, W1p, hf, psum, psq);
  reduce_stats <<<1, CH, 0, stream>>>(psum, psq, ssum, ssq, NBLK);
  bn_relu_bf16 <<<elemGrid, 256, 0, stream>>>(hf, ssum, ssq, g1, b1, h1b);

  // layer 2: conv -> BN stats -> BN + residual + ReLU (fp32 out)
  spconv       <<<NBLK, 256, 0, stream>>>(h1b, n2, W2p, hf, psum, psq);
  reduce_stats <<<1, CH, 0, stream>>>(psum, psq, ssum, ssq, NBLK);
  bn_res_relu  <<<elemGrid, 256, 0, stream>>>(hf, ssum, ssq, g2, b2, x, out);
}
